// Transformer_36902359007754
// MI455X (gfx1250) — matrix operand order repaired and checked by reading, not compiled, not run on hardware
//
#include <hip/hip_runtime.h>
#include <hip/hip_bf16.h>

typedef __bf16 bf16;
typedef unsigned int u32;
typedef __attribute__((ext_vector_type(16))) __bf16 v16bf;
typedef __attribute__((ext_vector_type(8)))  float  v8f;
typedef __attribute__((ext_vector_type(8)))  u32    v8u;

namespace {
constexpr int V_ = 32000, E_ = 1024, H_ = 16, DH_ = 64, DFF_ = 4096;
constexpr int B_ = 2, S_ = 512, T_ = B_ * S_;
constexpr float NEGV = -1e10f;
}

// ---------------------------------------------------------------------------
// Generic batched bf16 WMMA GEMM:  C[z] = act(alpha * A[z] @ Bt[z]^T + bias + res)
// A: [M,K] row-major bf16 (row stride lda), Bt: [N,K] row-major bf16 (B transposed).
// Register blocking: each wave computes a 32(M) x 64(N) tile as 2x4 sub-tiles of
// 16x16 via v_wmma_f32_16x16x32_bf16 (K stepped by 32). 128-thr block = 32x256 C.
// Requires: M % 32 == 0, N % 64 == 0, K % 32 == 0 (true for all call sites).
// ---------------------------------------------------------------------------
__global__ __launch_bounds__(128)
void gemm_wmma_kernel(const bf16* __restrict__ A, int lda, long long sA,
                      const bf16* __restrict__ Bt, int ldb, long long sB,
                      const float* __restrict__ bias, long long sBias,
                      const float* __restrict__ res, int ldr, long long sR,
                      float* __restrict__ Cf, int ldc, long long sCf,
                      bf16* __restrict__ Cb, int ldcb, long long sCb,
                      int M, int N, int K, float alpha, int relu, int transC)
{
    const int z = blockIdx.z;
    A  += (long long)z * sA;
    Bt += (long long)z * sB;
    const float* biasZ = bias ? bias + (long long)z * sBias : nullptr;
    const float* resZ  = res  ? res  + (long long)z * sR    : nullptr;
    float* CfZ = Cf ? Cf + (long long)z * sCf : nullptr;
    bf16*  CbZ = Cb ? Cb + (long long)z * sCb : nullptr;

    const int wave = threadIdx.x >> 5;
    const int lane = threadIdx.x & 31;
    const int n0 = blockIdx.x * 256 + wave * 64;
    const int m0 = blockIdx.y * 32;
    if (n0 >= N) return;   // wave-uniform: EXEC stays all-ones for WMMA

    const int r  = lane & 15;
    const int hi = lane >> 4;

    // A frag (contiguous k order k = 16*hi + s, matched by B): lane row = base+r; 16 contiguous K at hi*16.
    const bf16* aRow0 = A + (long long)(m0 + r)      * lda + hi * 16;
    const bf16* aRow1 = A + (long long)(m0 + 16 + r) * lda + hi * 16;
    // B frag (contiguous k order k = 16*hi + s, matched by A): lane col = base+r; 16 contiguous K at hi*16.
    const bf16* bRow0 = Bt + (long long)(n0 + r)      * ldb + hi * 16;
    const bf16* bRow1 = Bt + (long long)(n0 + 16 + r) * ldb + hi * 16;
    const bf16* bRow2 = Bt + (long long)(n0 + 32 + r) * ldb + hi * 16;
    const bf16* bRow3 = Bt + (long long)(n0 + 48 + r) * ldb + hi * 16;

    v8f acc[2][4];
    #pragma unroll
    for (int i = 0; i < 2; ++i)
        #pragma unroll
        for (int j = 0; j < 4; ++j)
            acc[i][j] = v8f{0.f,0.f,0.f,0.f,0.f,0.f,0.f,0.f};

    auto loadA = [&](const bf16* p) -> v16bf {
        const uint4 x0 = *(const uint4*)(p);
        const uint4 x1 = *(const uint4*)(p + 8);
        v8u u;
        u[0]=x0.x; u[1]=x0.y; u[2]=x0.z; u[3]=x0.w;
        u[4]=x1.x; u[5]=x1.y; u[6]=x1.z; u[7]=x1.w;
        return __builtin_bit_cast(v16bf, u);
    };
    auto loadB = [&](const bf16* p) -> v16bf {
        const uint4 x0 = *(const uint4*)(p);
        const uint4 x1 = *(const uint4*)(p + 8);
        v8u u;
        u[0]=x0.x; u[1]=x0.y; u[2]=x0.z; u[3]=x0.w;
        u[4]=x1.x; u[5]=x1.y; u[6]=x1.z; u[7]=x1.w;
        return __builtin_bit_cast(v16bf, u);
    };

    for (int k0 = 0; k0 < K; k0 += 32) {
        const v16bf a0 = loadA(aRow0 + k0);
        const v16bf a1 = loadA(aRow1 + k0);
        const v16bf b0 = loadB(bRow0 + k0);
        const v16bf b1 = loadB(bRow1 + k0);
        const v16bf b2 = loadB(bRow2 + k0);
        const v16bf b3 = loadB(bRow3 + k0);
        acc[0][0] = __builtin_amdgcn_wmma_f32_16x16x32_bf16(false, a0, false, b0, (short)0, acc[0][0], false, false);
        acc[0][1] = __builtin_amdgcn_wmma_f32_16x16x32_bf16(false, a0, false, b1, (short)0, acc[0][1], false, false);
        acc[0][2] = __builtin_amdgcn_wmma_f32_16x16x32_bf16(false, a0, false, b2, (short)0, acc[0][2], false, false);
        acc[0][3] = __builtin_amdgcn_wmma_f32_16x16x32_bf16(false, a0, false, b3, (short)0, acc[0][3], false, false);
        acc[1][0] = __builtin_amdgcn_wmma_f32_16x16x32_bf16(false, a1, false, b0, (short)0, acc[1][0], false, false);
        acc[1][1] = __builtin_amdgcn_wmma_f32_16x16x32_bf16(false, a1, false, b1, (short)0, acc[1][1], false, false);
        acc[1][2] = __builtin_amdgcn_wmma_f32_16x16x32_bf16(false, a1, false, b2, (short)0, acc[1][2], false, false);
        acc[1][3] = __builtin_amdgcn_wmma_f32_16x16x32_bf16(false, a1, false, b3, (short)0, acc[1][3], false, false);
    }

    // C layout per 16x16 tile: n = r, m = hi*8 + v (v = VGPR index 0..7)
    #pragma unroll
    for (int i = 0; i < 2; ++i) {
        #pragma unroll
        for (int j = 0; j < 4; ++j) {
            const int n = n0 + j * 16 + r;
            #pragma unroll
            for (int v = 0; v < 8; ++v) {
                const int m = m0 + i * 16 + hi * 8 + v;
                float x = acc[i][j][v] * alpha;
                if (biasZ) x += biasZ[n];
                if (resZ)  x += resZ[(long long)m * ldr + n];
                if (relu)  x = fmaxf(x, 0.f);
                if (CfZ) CfZ[(long long)m * ldc + n] = x;
                if (CbZ) {
                    if (transC) CbZ[(long long)n * ldcb + m] = (bf16)x;
                    else        CbZ[(long long)m * ldcb + n] = (bf16)x;
                }
            }
        }
    }
}

// ---------------------------------------------------------------------------
// Embedding gather + pad mask + positional encoding (base = max_len = 512).
// ---------------------------------------------------------------------------
__global__ __launch_bounds__(256)
void embed_kernel(const int* __restrict__ tok, const float* __restrict__ emb,
                  float* __restrict__ xf, bf16* __restrict__ xb,
                  int S, int E, float base)
{
    const int t = blockIdx.x;
    const int s = t % S;
    const int id = tok[t];
    const float m = (id != 0) ? 1.f : 0.f;
    const float pos = (float)(s + 1);
    for (int e = threadIdx.x; e < E; e += blockDim.x) {
        float w  = emb[(long long)id * E + e];
        float eq = powf(base, 2.f * (float)e / (float)E);
        float ang = pos / eq;
        float pe = (e & 1) ? cosf(ang) : sinf(ang);
        float v = (w + pe) * m;
        xf[(long long)t * E + e] = v;
        xb[(long long)t * E + e] = (bf16)v;
    }
}

// ---------------------------------------------------------------------------
// Weight f32 [K,N] -> bf16 transposed [N,K], LDS-tiled 32x32 for coalescing.
// Grid: (N/32, K/32, Z); block 256 = 32x8.
// ---------------------------------------------------------------------------
__global__ __launch_bounds__(256)
void transpose_bf16_kernel(const float* __restrict__ W, bf16* __restrict__ Wt,
                           int K, int N)
{
    __shared__ float tile[32][33];
    const long long z  = blockIdx.z;
    const long long KN = (long long)K * N;
    W  += z * KN;  Wt += z * KN;
    const int n0 = blockIdx.x * 32, k0 = blockIdx.y * 32;
    const int tx = threadIdx.x & 31, ty = threadIdx.x >> 5;   // 32 x 8
    #pragma unroll
    for (int i = 0; i < 32; i += 8) {
        const int k = k0 + ty + i, n = n0 + tx;
        if (k < K && n < N) tile[ty + i][tx] = W[(long long)k * N + n];
    }
    __syncthreads();
    #pragma unroll
    for (int i = 0; i < 32; i += 8) {
        const int n = n0 + ty + i, k = k0 + tx;
        if (n < N && k < K) Wt[(long long)n * K + k] = (bf16)tile[tx][ty + i];
    }
}

__global__ __launch_bounds__(256)
void convert_bf16_kernel(const float* __restrict__ a, bf16* __restrict__ o, long long n)
{
    long long i = (long long)blockIdx.x * blockDim.x + threadIdx.x;
    const long long stride = (long long)gridDim.x * blockDim.x;
    for (; i < n; i += stride) o[i] = (bf16)a[i];
}

// ---------------------------------------------------------------------------
// LayerNorm over E per token (input already contains residual), f32 + bf16 out.
// ---------------------------------------------------------------------------
__global__ __launch_bounds__(256)
void layernorm_kernel(const float* __restrict__ x, const float* __restrict__ g,
                      const float* __restrict__ beta,
                      float* __restrict__ of, bf16* __restrict__ ob, int E)
{
    const int t = blockIdx.x, tid = threadIdx.x;
    const float* xr = x + (long long)t * E;
    float s = 0.f, ss = 0.f;
    for (int e = tid; e < E; e += 256) { float v = xr[e]; s += v; ss += v * v; }
    for (int o = 16; o; o >>= 1) { s += __shfl_down(s, o, 32); ss += __shfl_down(ss, o, 32); }
    __shared__ float shs[8], shq[8];
    const int w = tid >> 5, l = tid & 31;
    if (l == 0) { shs[w] = s; shq[w] = ss; }
    __syncthreads();
    float ts = 0.f, tq = 0.f;
    for (int i = 0; i < 8; ++i) { ts += shs[i]; tq += shq[i]; }
    const float mu = ts / E;
    const float var = tq / E - mu * mu;
    const float rstd = rsqrtf(var + 1e-5f);
    for (int e = tid; e < E; e += 256) {
        float v = (xr[e] - mu) * rstd * g[e] + beta[e];
        if (of) of[(long long)t * E + e] = v;
        if (ob) ob[(long long)t * E + e] = (bf16)v;
    }
}

// ---------------------------------------------------------------------------
// Masked (query-pad) + causal softmax; f32 scores -> bf16 probs.
// scores layout [B][H][Sq][Skv]; blockIdx.y = b*H + h.
// ---------------------------------------------------------------------------
__global__ __launch_bounds__(256)
void softmax_kernel(const float* __restrict__ sc, bf16* __restrict__ pr,
                    const int* __restrict__ tok, int Sq, int Skv, int Hh, int causal)
{
    const int q = blockIdx.x;
    const int z = blockIdx.y;
    const int b = z / Hh;
    const float* row = sc + ((long long)z * Sq + q) * Skv;
    bf16* orow = pr + ((long long)z * Sq + q) * Skv;
    const bool padq = (tok[b * Sq + q] == 0);
    const int tid = threadIdx.x;

    float mx = -3.4e38f;
    for (int k = tid; k < Skv; k += 256) {
        float v = padq ? NEGV : row[k];
        if (causal && k > q) v += NEGV;
        mx = fmaxf(mx, v);
    }
    for (int o = 16; o; o >>= 1) mx = fmaxf(mx, __shfl_down(mx, o, 32));
    __shared__ float shm[8], shsum[8];
    const int w = tid >> 5, l = tid & 31;
    if (l == 0) shm[w] = mx;
    __syncthreads();
    float gmx = shm[0];
    for (int i = 1; i < 8; ++i) gmx = fmaxf(gmx, shm[i]);

    float sum = 0.f;
    for (int k = tid; k < Skv; k += 256) {
        float v = padq ? NEGV : row[k];
        if (causal && k > q) v += NEGV;
        sum += expf(v - gmx);
    }
    for (int o = 16; o; o >>= 1) sum += __shfl_down(sum, o, 32);
    if (l == 0) shsum[w] = sum;
    __syncthreads();
    float gs = 0.f;
    for (int i = 0; i < 8; ++i) gs += shsum[i];
    const float inv = 1.f / gs;
    for (int k = tid; k < Skv; k += 256) {
        float v = padq ? NEGV : row[k];
        if (causal && k > q) v += NEGV;
        orow[k] = (bf16)(expf(v - gmx) * inv);
    }
}

// ============================ host orchestration ============================

namespace {

struct AttnP  { const float *Wq,*bq,*Wk,*bk,*Wv,*bv,*Wo,*bo; };
struct BlockP { const float *Wq,*bq,*Wk,*bk,*Wv,*bv,*Wo,*bo,
                            *g1,*be1,*W1,*b1,*W2,*b2,*g2,*be2; };

struct Ctx {
    hipStream_t st;
    bf16 *emb_bf, *wq_t, *wk_t, *wv_t, *wo_t, *w1_t, *w2_t;
    bf16 *x_bf, *y_bf, *n_bf, *ffn_bf, *o_bf, *qb, *kb, *vt, *probs;
    float *x_f, *y_f, *x2, *n_f, *h2, *scores;
};

void gemm(hipStream_t st,
          const bf16* A, int lda, long long sA,
          const bf16* Bt, int ldb, long long sB,
          const float* bias, long long sBias,
          const float* res, int ldr, long long sR,
          float* Cf, int ldc, long long sCf,
          bf16* Cb, int ldcb, long long sCb,
          int M, int N, int K, int Z,
          float alpha = 1.f, int relu = 0, int transC = 0)
{
    dim3 g((N + 255) / 256, M / 32, Z);
    gemm_wmma_kernel<<<g, 128, 0, st>>>(A, lda, sA, Bt, ldb, sB, bias, sBias,
                                        res, ldr, sR, Cf, ldc, sCf, Cb, ldcb, sCb,
                                        M, N, K, alpha, relu, transC);
}

void transposeW(hipStream_t st, const float* W, bf16* Wt, int K, int N, int Z)
{
    dim3 g(N / 32, K / 32, Z);
    transpose_bf16_kernel<<<g, 256, 0, st>>>(W, Wt, K, N);
}

// Multi-head attention: Q from qs, K/V from kvs, residual (f32) added in Wo epilogue.
void run_mha(Ctx& c, const bf16* qs, const bf16* kvs, const float* resid,
             const int* qtok, const AttnP& p, int causal, float* out)
{
    const int headKN = DH_ * DH_;
    transposeW(c.st, p.Wq, c.wq_t, DH_, DH_, H_);
    transposeW(c.st, p.Wk, c.wk_t, DH_, DH_, H_);
    transposeW(c.st, p.Wv, c.wv_t, DH_, DH_, H_);
    transposeW(c.st, p.Wo, c.wo_t, E_, E_, 1);

    for (int b = 0; b < B_; ++b) {
        const bf16* xq  = qs  + (long long)b * S_ * E_;
        const bf16* xkv = kvs + (long long)b * S_ * E_;
        // Q,K: [b][h][s][d]; V stored transposed: [b][h][d][s]
        gemm(c.st, xq,  E_, DH_, c.wq_t, DH_, headKN, p.bq, DH_, nullptr, 0, 0,
             nullptr, 0, 0, c.qb + (long long)b * H_ * S_ * DH_, DH_, (long long)S_ * DH_,
             S_, DH_, DH_, H_);
        gemm(c.st, xkv, E_, DH_, c.wk_t, DH_, headKN, p.bk, DH_, nullptr, 0, 0,
             nullptr, 0, 0, c.kb + (long long)b * H_ * S_ * DH_, DH_, (long long)S_ * DH_,
             S_, DH_, DH_, H_);
        gemm(c.st, xkv, E_, DH_, c.wv_t, DH_, headKN, p.bv, DH_, nullptr, 0, 0,
             nullptr, 0, 0, c.vt + (long long)b * H_ * DH_ * S_, S_, (long long)DH_ * S_,
             S_, DH_, DH_, H_, 1.f, 0, /*transC=*/1);
        // scores = Q @ K^T / 8  (Bt = K rows directly)
        gemm(c.st, c.qb + (long long)b * H_ * S_ * DH_, DH_, (long long)S_ * DH_,
             c.kb + (long long)b * H_ * S_ * DH_, DH_, (long long)S_ * DH_,
             nullptr, 0, nullptr, 0, 0,
             c.scores + (long long)b * H_ * S_ * S_, S_, (long long)S_ * S_,
             nullptr, 0, 0, S_, S_, DH_, H_, 0.125f);
    }
    softmax_kernel<<<dim3(S_, B_ * H_), 256, 0, c.st>>>(c.scores, c.probs, qtok, S_, S_, H_, causal);
    for (int b = 0; b < B_; ++b) {
        // O_head = P @ V  (Bt = V^T rows), heads interleaved into [s, h*64+d]
        gemm(c.st, c.probs + (long long)b * H_ * S_ * S_, S_, (long long)S_ * S_,
             c.vt + (long long)b * H_ * DH_ * S_, S_, (long long)DH_ * S_,
             nullptr, 0, nullptr, 0, 0, nullptr, 0, 0,
             c.o_bf + (long long)b * S_ * E_, E_, DH_,
             S_, DH_, S_, H_);
    }
    // out = O @ Wo + bo + resid
    gemm(c.st, c.o_bf, E_, 0, c.wo_t, E_, 0, p.bo, 0, resid, E_, 0,
         out, E_, 0, nullptr, 0, 0, T_, E_, E_, 1);
}

void run_block(Ctx& c, bf16* x_bf, float* x_f, const bf16* kv_bf,
               const int* qtok, int causal, const BlockP& p)
{
    AttnP a{p.Wq, p.bq, p.Wk, p.bk, p.Wv, p.bv, p.Wo, p.bo};
    run_mha(c, x_bf, kv_bf, x_f, qtok, a, causal, c.x2);
    layernorm_kernel<<<T_, 256, 0, c.st>>>(c.x2, p.g1, p.be1, c.n_f, c.n_bf, E_);
    transposeW(c.st, p.W1, c.w1_t, E_, DFF_, 1);
    gemm(c.st, c.n_bf, E_, 0, c.w1_t, E_, 0, p.b1, 0, nullptr, 0, 0,
         nullptr, 0, 0, c.ffn_bf, DFF_, 0, T_, DFF_, E_, 1, 1.f, /*relu=*/1);
    transposeW(c.st, p.W2, c.w2_t, DFF_, E_, 1);
    gemm(c.st, c.ffn_bf, DFF_, 0, c.w2_t, DFF_, 0, p.b2, 0, c.n_f, E_, 0,
         c.h2, E_, 0, nullptr, 0, 0, T_, E_, DFF_, 1);
    layernorm_kernel<<<T_, 256, 0, c.st>>>(c.h2, p.g2, p.be2, x_f, x_bf, E_);
}

// jax-tree (alphabetical) flatten of setup_inputs():
// 0:batch 1:target 2:params.b 3..18:dec{W1,W2,Wk,Wo,Wq,Wv,b1,b2,be1,be2,bk,bo,bq,bv,g1,g2}
// 19:emb 20..35:enc{...same...} 36:params.g 37..44:ma{Wk,Wo,Wq,Wv,bk,bo,bq,bv}
BlockP get_block(void* const* din, int base, int layer)
{
    auto f = [&](int leaf, long long per) {
        return (const float*)din[base + leaf] + (long long)layer * per;
    };
    BlockP q;
    q.W1 = f(0, (long long)E_ * DFF_);   q.W2 = f(1, (long long)DFF_ * E_);
    q.Wk = f(2, (long long)H_ * DH_ * DH_); q.Wo = f(3, (long long)E_ * E_);
    q.Wq = f(4, (long long)H_ * DH_ * DH_); q.Wv = f(5, (long long)H_ * DH_ * DH_);
    q.b1 = f(6, DFF_);  q.b2 = f(7, E_);  q.be1 = f(8, E_);  q.be2 = f(9, E_);
    q.bk = f(10, (long long)H_ * DH_);  q.bo = f(11, E_);
    q.bq = f(12, (long long)H_ * DH_);  q.bv = f(13, (long long)H_ * DH_);
    q.g1 = f(14, E_);  q.g2 = f(15, E_);
    return q;
}

} // namespace

extern "C" void kernel_launch(void* const* d_in, const int* in_sizes, int n_in,
                              void* d_out, int out_size, void* d_ws, size_t ws_size,
                              hipStream_t stream)
{
    (void)in_sizes; (void)out_size; (void)ws_size;
    if (n_in < 45) return;

    const int*   batch  = (const int*)d_in[0];
    const int*   target = (const int*)d_in[1];
    const float* p_b    = (const float*)d_in[2];
    const float* emb    = (const float*)d_in[19];
    const float* p_g    = (const float*)d_in[36];
    const AttnP  ma{ (const float*)d_in[39], (const float*)d_in[43],   // Wq,bq
                     (const float*)d_in[37], (const float*)d_in[41],   // Wk,bk
                     (const float*)d_in[40], (const float*)d_in[44],   // Wv,bv
                     (const float*)d_in[38], (const float*)d_in[42] }; // Wo,bo

    char* bp = (char*)d_ws;
    auto alloc = [&](size_t bytes) -> void* {
        void* r = bp; bp += (bytes + 255) & ~(size_t)255; return r;
    };

    Ctx c; c.st = stream;
    c.emb_bf = (bf16*)alloc((size_t)V_ * E_ * 2);
    c.wq_t   = (bf16*)alloc((size_t)H_ * DH_ * DH_ * 2);
    c.wk_t   = (bf16*)alloc((size_t)H_ * DH_ * DH_ * 2);
    c.wv_t   = (bf16*)alloc((size_t)H_ * DH_ * DH_ * 2);
    c.wo_t   = (bf16*)alloc((size_t)E_ * E_ * 2);
    c.w1_t   = (bf16*)alloc((size_t)E_ * DFF_ * 2);
    c.w2_t   = (bf16*)alloc((size_t)DFF_ * E_ * 2);
    c.x_f    = (float*)alloc((size_t)T_ * E_ * 4);
    c.y_f    = (float*)alloc((size_t)T_ * E_ * 4);
    c.x2     = (float*)alloc((size_t)T_ * E_ * 4);
    c.n_f    = (float*)alloc((size_t)T_ * E_ * 4);
    c.h2     = (float*)alloc((size_t)T_ * E_ * 4);
    c.scores = (float*)alloc((size_t)B_ * H_ * S_ * S_ * 4);
    c.x_bf   = (bf16*)alloc((size_t)T_ * E_ * 2);
    c.y_bf   = (bf16*)alloc((size_t)T_ * E_ * 2);
    c.n_bf   = (bf16*)alloc((size_t)T_ * E_ * 2);
    c.o_bf   = (bf16*)alloc((size_t)T_ * E_ * 2);
    c.ffn_bf = (bf16*)alloc((size_t)T_ * DFF_ * 2);
    c.qb     = (bf16*)alloc((size_t)B_ * H_ * S_ * DH_ * 2);
    c.kb     = (bf16*)alloc((size_t)B_ * H_ * S_ * DH_ * 2);
    c.vt     = (bf16*)alloc((size_t)B_ * H_ * DH_ * S_ * 2);
    c.probs  = (bf16*)alloc((size_t)B_ * H_ * S_ * S_ * 2);

    // Tied-embedding weights once (bf16 [V,E] == Bt for the logits GEMM; fits in L2)
    convert_bf16_kernel<<<2048, 256, 0, stream>>>(emb, c.emb_bf, (long long)V_ * E_);

    // ---- encoder ----
    embed_kernel<<<T_, 256, 0, stream>>>(batch, emb, c.x_f, c.x_bf, S_, E_, 512.f);
    run_block(c, c.x_bf, c.x_f, c.x_bf, batch, 0, get_block(d_in, 20, 0));
    run_block(c, c.x_bf, c.x_f, c.x_bf, batch, 0, get_block(d_in, 20, 1));

    // ---- decoder ----
    embed_kernel<<<T_, 256, 0, stream>>>(target, emb, c.y_f, c.y_bf, S_, E_, 512.f);
    run_mha(c, c.y_bf, c.y_bf, c.y_f, target, ma, /*causal=*/1, c.x2);
    layernorm_kernel<<<T_, 256, 0, stream>>>(c.x2, p_g, p_b, c.y_f, c.y_bf, E_);
    run_block(c, c.y_bf, c.y_f, c.x_bf /*enc out*/, target, 0, get_block(d_in, 3, 0));
    run_block(c, c.y_bf, c.y_f, c.y_bf,             target, 0, get_block(d_in, 3, 1));

    // ---- tied-embedding logits: [T,E] @ emb^T -> [T,V] f32 ----
    gemm(stream, c.y_bf, E_, 0, c.emb_bf, E_, 0, nullptr, 0, nullptr, 0, 0,
         (float*)d_out, V_, 0, nullptr, 0, 0, T_, V_, E_, 1);
}
